// Decoder_75849122448079
// MI455X (gfx1250) — matrix operand order repaired and checked by reading, not compiled, not run on hardware
//
#include <hip/hip_runtime.h>

typedef __bf16 bf16;
typedef __attribute__((ext_vector_type(16))) bf16  v16bf;
typedef __attribute__((ext_vector_type(8)))  bf16  v8bf;
typedef __attribute__((ext_vector_type(8)))  float v8f;
typedef __attribute__((ext_vector_type(4)))  unsigned int u32x4;
typedef __attribute__((ext_vector_type(8)))  int i32x8;
typedef __attribute__((ext_vector_type(4)))  int i32x4;

#define N_SEQ 4096
#define WDIM  512
#define PDIM  64
#define HEADS 8
#define FDIM  2048

#if defined(__gfx1250__) && __has_builtin(__builtin_amdgcn_tensor_load_to_lds)
#define HAVE_TDM 1
#else
#define HAVE_TDM 0
#endif

// ------------------------------------------------------------------
// elementwise helpers
// ------------------------------------------------------------------
__global__ void cvt_f32_bf16(const float* __restrict__ src, bf16* __restrict__ dst, int n) {
  int i = blockIdx.x * blockDim.x + threadIdx.x;
  if (i < n) dst[i] = (bf16)src[i];
}

// convert + transpose: src f32 [R,C] -> dst bf16 [C,R]
__global__ void cvt_transpose_f32_bf16(const float* __restrict__ src, bf16* __restrict__ dst,
                                       int R, int C) {
  int idx = blockIdx.x * blockDim.x + threadIdx.x;
  if (idx >= R * C) return;
  int r = idx / C, c = idx - r * C;
  dst[(size_t)c * R + r] = (bf16)src[idx];
}

// w_o_sum^T[j][i] = sum_h w_o[h*P + i][j]  (collapsed concat, stored transposed)
__global__ void wo_sumT_kernel(const float* __restrict__ wo, bf16* __restrict__ outT) {
  int idx = blockIdx.x * blockDim.x + threadIdx.x; // < 64*512
  int i = idx >> 9, j = idx & 511;                 // i: P index, j: W index
  float s = 0.f;
#pragma unroll
  for (int h = 0; h < HEADS; ++h) s += wo[(size_t)(h * PDIM + i) * WDIM + j];
  outT[(size_t)j * PDIM + i] = (bf16)s;
}

__global__ void transpose_bf16(const bf16* __restrict__ in, bf16* __restrict__ out, int R, int C) {
  int idx = blockIdx.x * blockDim.x + threadIdx.x;
  if (idx >= R * C) return;
  int r = idx / C, c = idx - r * C;
  out[(size_t)c * R + r] = in[idx];
}

// ------------------------------------------------------------------
// TDM staging: one descriptor copies a (tile_d1 rows x 32 elems) bf16
// tile (row-major, row stride = rowlen elements) into LDS, contiguous.
// ------------------------------------------------------------------
#if HAVE_TDM
__device__ __forceinline__ unsigned lds_byte_offset(const void* p) {
  typedef __attribute__((address_space(3))) const void* lds_cptr;
  return (unsigned)(unsigned long long)(lds_cptr)p;
}

__device__ __forceinline__ void tdm_load_tile(const bf16* gsrc, unsigned lds_off,
                                              int rowlen, int rows_total, int tile_d1) {
  unsigned long long ga = (unsigned long long)(size_t)gsrc;
  u32x4 g0;
  g0.x = 1u;                                                  // count=1, no gather
  g0.y = lds_off;                                             // lds_addr (bytes)
  g0.z = (unsigned)(ga & 0xffffffffu);                        // global_addr[31:0]
  g0.w = (unsigned)((ga >> 32) & 0x01ffffffu) | (2u << 30);   // addr[56:32] | type=2
  i32x8 g1;
  g1[0] = (int)(1u << 16);                                    // wg_mask=0, data_size=1 (2B)
  g1[1] = (int)(((unsigned)rowlen & 0xffffu) << 16);          // tensor_dim0[15:0]
  g1[2] = (int)((((unsigned)rowlen >> 16) & 0xffffu) |
                (((unsigned)rows_total & 0xffffu) << 16));    // dim0 hi | dim1 lo
  g1[3] = (int)((((unsigned)rows_total >> 16) & 0xffffu) |
                (32u << 16));                                 // dim1 hi | tile_dim0=32
  g1[4] = tile_d1;                                            // tile_dim1, tile_dim2=0
  g1[5] = rowlen;                                             // tensor_dim0_stride lo
  g1[6] = 0;
  g1[7] = 0;
  i32x4 z4 = {0, 0, 0, 0};
#if __has_include(<hip/amd_detail/amd_gfx1250_TDM.h>)
  i32x8 z8 = {0, 0, 0, 0, 0, 0, 0, 0};
  __builtin_amdgcn_tensor_load_to_lds(g0, g1, z4, z4, z8, 0);
#else
  __builtin_amdgcn_tensor_load_to_lds(g0, g1, z4, z4, 0);
#endif
}

__device__ __forceinline__ void tdm_wait0() {
#if __has_builtin(__builtin_amdgcn_s_wait_tensorcnt)
  __builtin_amdgcn_s_wait_tensorcnt(0);
#else
  asm volatile("s_wait_tensorcnt 0x0" ::: "memory");
#endif
}
#endif  // HAVE_TDM

// ------------------------------------------------------------------
// bf16 WMMA GEMM:  C[M,N] = act( scale * (A[M,K] @ Bt^T[K,N]) + bias )
// A row-major [M,K] bf16; Bt row-major [N,K] bf16 (i.e. B col-major).
// C f32 or bf16.  M%128==0, N%64==0, K%32==0.
// ------------------------------------------------------------------
#define BM 128
#define BN 64
#define BK 32

__device__ __forceinline__ void mma_tile(const bf16* __restrict__ As, const bf16* __restrict__ Bs,
                                         int wm, int wn, int l16, int lh, v8f (&acc)[2][2]) {
  // A fragment: lane row = l16, VGPR0-3 hold k = lh*8 .. +7, VGPR4-7 hold k = lh*8+16 .. +7
  v16bf af[2], bfg[2];
#pragma unroll
  for (int mi = 0; mi < 2; ++mi) {
    int r = wm * 32 + mi * 16 + l16;
    v8bf p0 = *reinterpret_cast<const v8bf*>(&As[r * BK + lh * 8]);
    v8bf p1 = *reinterpret_cast<const v8bf*>(&As[r * BK + lh * 8 + 16]);
    af[mi]  = __builtin_shufflevector(p0, p1, 0,1,2,3,4,5,6,7,8,9,10,11,12,13,14,15);
  }
  // B fragment: lane col = l16, VGPR0-3 hold k = lh*8 .. +7, VGPR4-7 hold k = lh*8+16 .. +7 ([n][k] LDS layout)
#pragma unroll
  for (int ni = 0; ni < 2; ++ni) {
    int n = wn * 32 + ni * 16 + l16;
    v8bf p0 = *reinterpret_cast<const v8bf*>(&Bs[n * BK + lh * 8]);
    v8bf p1 = *reinterpret_cast<const v8bf*>(&Bs[n * BK + lh * 8 + 16]);
    bfg[ni] = __builtin_shufflevector(p0, p1, 0,1,2,3,4,5,6,7,8,9,10,11,12,13,14,15);
  }
#pragma unroll
  for (int mi = 0; mi < 2; ++mi)
#pragma unroll
    for (int ni = 0; ni < 2; ++ni)
      acc[mi][ni] = __builtin_amdgcn_wmma_f32_16x16x32_bf16(
          false, af[mi], false, bfg[ni], (short)0, acc[mi][ni], false, false);
}

template <bool OUT_BF16>
__global__ __launch_bounds__(256) void gemm_bf16_wmma(
    const bf16* __restrict__ A, const bf16* __restrict__ Bt, void* __restrict__ Cout,
    int M, int N, int K, float scale, const float* __restrict__ bias, int leaky) {
  const int tid  = threadIdx.x;
  const int lane = tid & 31;
  const int wave = tid >> 5;
  const int wm   = wave >> 1;   // 0..3 -> 32-row slice
  const int wn   = wave & 1;    // 0..1 -> 32-col slice
  const int m0   = blockIdx.y * BM;
  const int n0   = blockIdx.x * BN;
  const int l16  = lane & 15;
  const int lh   = lane >> 4;

  v8f acc[2][2];
#pragma unroll
  for (int a = 0; a < 2; ++a)
#pragma unroll
    for (int b = 0; b < 2; ++b) acc[a][b] = (v8f){0.f,0.f,0.f,0.f,0.f,0.f,0.f,0.f};

#if HAVE_TDM
  // double-buffered LDS; tiles staged exclusively by the Tensor Data Mover
  __shared__ __align__(16) bf16 As[2][BM * BK];
  __shared__ __align__(16) bf16 Bs[2][BN * BK];
  const bf16* agbase = &A[(size_t)m0 * K];
  const bf16* bgbase = &Bt[(size_t)n0 * K];

  if (wave == 0) {
    tdm_load_tile(agbase, lds_byte_offset(&As[0][0]), K, M - m0, BM);
    tdm_load_tile(bgbase, lds_byte_offset(&Bs[0][0]), K, N - n0, BN);
    tdm_wait0();
  }
  __syncthreads();

  int cur = 0;
  for (int k0 = 0; k0 < K; k0 += BK) {
    int nxt = cur ^ 1;
    if (k0 + BK < K && wave == 0) {   // prefetch next tile pair while we compute
      tdm_load_tile(agbase + (k0 + BK), lds_byte_offset(&As[nxt][0]), K, M - m0, BM);
      tdm_load_tile(bgbase + (k0 + BK), lds_byte_offset(&Bs[nxt][0]), K, N - n0, BN);
    }
    mma_tile(&As[cur][0], &Bs[cur][0], wm, wn, l16, lh, acc);
    if (wave == 0) tdm_wait0();
    __syncthreads();
    cur = nxt;
  }
#else
  // fallback: vector-load staging (both tiles contiguous thanks to Bt layout)
  __shared__ __align__(16) bf16 As[BM * BK];
  __shared__ __align__(16) bf16 Bs[BN * BK];
  for (int k0 = 0; k0 < K; k0 += BK) {
#pragma unroll
    for (int it = 0; it < 2; ++it) {
      int idx = tid + it * 256;
      int row = idx >> 2;
      int c8  = (idx & 3) << 3;
      *reinterpret_cast<uint4*>(&As[row * BK + c8]) =
          *reinterpret_cast<const uint4*>(&A[(size_t)(m0 + row) * K + k0 + c8]);
    }
    {
      int row = tid >> 2;
      int c8  = (tid & 3) << 3;
      *reinterpret_cast<uint4*>(&Bs[row * BK + c8]) =
          *reinterpret_cast<const uint4*>(&Bt[(size_t)(n0 + row) * K + k0 + c8]);
    }
    __syncthreads();
    mma_tile(As, Bs, wm, wn, l16, lh, acc);
    __syncthreads();
  }
#endif

  // epilogue: C 16x16 f32 layout -> lane col = l16, rows lh*8 + r
#pragma unroll
  for (int mi = 0; mi < 2; ++mi) {
#pragma unroll
    for (int ni = 0; ni < 2; ++ni) {
      int rbase = m0 + wm * 32 + mi * 16 + lh * 8;
      int col   = n0 + wn * 32 + ni * 16 + l16;
      float bv  = bias ? bias[col] : 0.f;
#pragma unroll
      for (int r = 0; r < 8; ++r) {
        float v = acc[mi][ni][r] * scale + bv;
        if (leaky) v = v > 0.f ? v : 0.01f * v;
        if (OUT_BF16)
          reinterpret_cast<bf16*>(Cout)[(size_t)(rbase + r) * N + col] = (bf16)v;
        else
          reinterpret_cast<float*>(Cout)[(size_t)(rbase + r) * N + col] = v;
      }
    }
  }
}

// ------------------------------------------------------------------
// reductions
// ------------------------------------------------------------------
__device__ __forceinline__ float wred_max(float v) {
#pragma unroll
  for (int o = 16; o > 0; o >>= 1) v = fmaxf(v, __shfl_xor(v, o, 32));
  return v;
}
__device__ __forceinline__ float wred_sum(float v) {
#pragma unroll
  for (int o = 16; o > 0; o >>= 1) v += __shfl_xor(v, o, 32);
  return v;
}

// in-place row softmax on bf16 scores (stats in f32)
__global__ __launch_bounds__(256) void softmax_rows_bf16(bf16* __restrict__ S, int n) {
  __shared__ float sm[8];
  bf16* p  = S + (size_t)blockIdx.x * n;
  int tid  = threadIdx.x, lane = tid & 31, w = tid >> 5;

  float mx = -1e30f;
  for (int i = tid; i < n; i += 256) mx = fmaxf(mx, (float)p[i]);
  mx = wred_max(mx);
  if (lane == 0) sm[w] = mx;
  __syncthreads();
  float m = sm[0];
#pragma unroll
  for (int i = 1; i < 8; ++i) m = fmaxf(m, sm[i]);
  __syncthreads();

  float s = 0.f;
  for (int i = tid; i < n; i += 256) {
    float e = __expf((float)p[i] - m);
    p[i] = (bf16)e;
    s += e;
  }
  s = wred_sum(s);
  if (lane == 0) sm[w] = s;
  __syncthreads();
  float tot = 0.f;
#pragma unroll
  for (int i = 0; i < 8; ++i) tot += sm[i];
  float inv = 1.f / tot;
  for (int i = tid; i < n; i += 256) p[i] = (bf16)((float)p[i] * inv);
}

// out = LayerNorm(a + b) * g + beta ; optional bf16 shadow copy for next GEMM
__global__ __launch_bounds__(256) void add_ln_kernel(
    const float* __restrict__ a, const float* __restrict__ b,
    const float* __restrict__ g, const float* __restrict__ bt,
    float* __restrict__ out, bf16* __restrict__ outb) {
  __shared__ float sm[8];
  int row = blockIdx.x, tid = threadIdx.x, lane = tid & 31, w = tid >> 5;
  const float* ar = a + (size_t)row * WDIM;
  const float* br = b + (size_t)row * WDIM;
  float x0 = ar[tid] + br[tid];
  float x1 = ar[tid + 256] + br[tid + 256];

  float s = wred_sum(x0 + x1);
  if (lane == 0) sm[w] = s;
  __syncthreads();
  float tot = 0.f;
#pragma unroll
  for (int i = 0; i < 8; ++i) tot += sm[i];
  float mean = tot * (1.f / WDIM);
  __syncthreads();

  float d0 = x0 - mean, d1 = x1 - mean;
  float vs = wred_sum(d0 * d0 + d1 * d1);
  if (lane == 0) sm[w] = vs;
  __syncthreads();
  float vtot = 0.f;
#pragma unroll
  for (int i = 0; i < 8; ++i) vtot += sm[i];
  float rstd = rsqrtf(vtot * (1.f / WDIM) + 1e-5f);

  float y0 = d0 * rstd * g[tid] + bt[tid];
  float y1 = d1 * rstd * g[tid + 256] + bt[tid + 256];
  float* orow = out + (size_t)row * WDIM;
  orow[tid] = y0;
  orow[tid + 256] = y1;
  if (outb) {
    bf16* ob = outb + (size_t)row * WDIM;
    ob[tid] = (bf16)y0;
    ob[tid + 256] = (bf16)y1;
  }
}

// ------------------------------------------------------------------
// host-side orchestration
// ------------------------------------------------------------------
extern "C" void kernel_launch(void* const* d_in, const int* in_sizes, int n_in,
                              void* d_out, int out_size, void* d_ws, size_t ws_size,
                              hipStream_t stream) {
  const float* x   = (const float*)d_in[0];
  const float* wqm = (const float*)d_in[1];
  const float* wkm = (const float*)d_in[2];
  const float* wvm = (const float*)d_in[3];
  const float* wqc = (const float*)d_in[4];
  const float* wkc = (const float*)d_in[5];
  const float* wvc = (const float*)d_in[6];
  const float* wo  = (const float*)d_in[7];
  const float* lng = (const float*)d_in[8];
  const float* lnb = (const float*)d_in[9];
  const float* w1  = (const float*)d_in[10];
  const float* b1  = (const float*)d_in[11];
  const float* w2  = (const float*)d_in[12];
  const float* b2  = (const float*)d_in[13];
  float* outp = (float*)d_out;

  size_t cur = 0;
  auto alloc = [&](size_t bytes) -> void* {
    void* p = (char*)d_ws + cur;
    cur += (bytes + 255) & ~(size_t)255;
    return p;
  };
  bf16* xb     = (bf16*)alloc((size_t)N_SEQ * WDIM * 2);
  bf16* wqmT   = (bf16*)alloc((size_t)WDIM * PDIM * 2);   // [P,W]
  bf16* wkmT   = (bf16*)alloc((size_t)WDIM * PDIM * 2);
  bf16* wvmT   = (bf16*)alloc((size_t)WDIM * PDIM * 2);
  bf16* wqcT   = (bf16*)alloc((size_t)WDIM * PDIM * 2);
  bf16* wkcT   = (bf16*)alloc((size_t)WDIM * PDIM * 2);
  bf16* wvcT   = (bf16*)alloc((size_t)WDIM * PDIM * 2);
  bf16* wosumT = (bf16*)alloc((size_t)WDIM * PDIM * 2);   // [W,P]
  bf16* w1T    = (bf16*)alloc((size_t)WDIM * FDIM * 2);   // [F,W]
  bf16* w2T    = (bf16*)alloc((size_t)FDIM * WDIM * 2);   // [W,F]
  bf16* qb     = (bf16*)alloc((size_t)N_SEQ * PDIM * 2);
  bf16* kb     = (bf16*)alloc((size_t)N_SEQ * PDIM * 2);  // K, used directly as Bt of QK^T
  bf16* vb     = (bf16*)alloc((size_t)N_SEQ * PDIM * 2);
  bf16* vTb    = (bf16*)alloc((size_t)PDIM * N_SEQ * 2);  // V^T, Bt of PV
  bf16* Sb     = (bf16*)alloc((size_t)N_SEQ * N_SEQ * 2);
  bf16* Ab     = (bf16*)alloc((size_t)N_SEQ * PDIM * 2);
  bf16* mmb    = (bf16*)alloc((size_t)N_SEQ * WDIM * 2);
  float* mhf   = (float*)alloc((size_t)N_SEQ * WDIM * 4);
  float* hf    = (float*)alloc((size_t)N_SEQ * WDIM * 4);
  bf16* hb     = (bf16*)alloc((size_t)N_SEQ * WDIM * 2);
  bf16* y1b    = (bf16*)alloc((size_t)N_SEQ * FDIM * 2);
  float* y2f   = (float*)alloc((size_t)N_SEQ * WDIM * 4);

  auto cvt = [&](const float* s, bf16* d, int n) {
    cvt_f32_bf16<<<(n + 255) / 256, 256, 0, stream>>>(s, d, n);
  };
  auto cvtT = [&](const float* s, bf16* d, int R, int C) {
    cvt_transpose_f32_bf16<<<(R * C + 255) / 256, 256, 0, stream>>>(s, d, R, C);
  };
  auto gemm = [&](const bf16* A, const bf16* Bt, void* C, int M, int Nn, int K,
                  float scale, const float* bias, int leaky, bool outbf) {
    dim3 grid(Nn / BN, M / BM);
    if (outbf)
      gemm_bf16_wmma<true><<<grid, 256, 0, stream>>>(A, Bt, C, M, Nn, K, scale, bias, leaky);
    else
      gemm_bf16_wmma<false><<<grid, 256, 0, stream>>>(A, Bt, C, M, Nn, K, scale, bias, leaky);
  };

  // --- precision staging (weights stored transposed = col-major B) ---
  cvt(x, xb, N_SEQ * WDIM);
  cvtT(wqm, wqmT, WDIM, PDIM);
  cvtT(wkm, wkmT, WDIM, PDIM);
  cvtT(wvm, wvmT, WDIM, PDIM);
  cvtT(wqc, wqcT, WDIM, PDIM);
  cvtT(wkc, wkcT, WDIM, PDIM);
  cvtT(wvc, wvcT, WDIM, PDIM);
  cvtT(w1, w1T, WDIM, FDIM);
  cvtT(w2, w2T, FDIM, WDIM);
  wo_sumT_kernel<<<(PDIM * WDIM) / 256, 256, 0, stream>>>(wo, wosumT);

  const float invsqrtP = 0.125f;  // 1/sqrt(64)

  // --- masked self-attention ---
  gemm(xb, wqmT, qb, N_SEQ, PDIM, WDIM, 1.f, nullptr, 0, true);
  gemm(xb, wkmT, kb, N_SEQ, PDIM, WDIM, 1.f, nullptr, 0, true);
  gemm(xb, wvmT, vb, N_SEQ, PDIM, WDIM, 1.f, nullptr, 0, true);
  transpose_bf16<<<(N_SEQ * PDIM) / 256, 256, 0, stream>>>(vb, vTb, N_SEQ, PDIM);
  gemm(qb, kb, Sb, N_SEQ, N_SEQ, PDIM, invsqrtP, nullptr, 0, true);  // Bt = K
  softmax_rows_bf16<<<N_SEQ, 256, 0, stream>>>(Sb, N_SEQ);
  gemm(Sb, vTb, Ab, N_SEQ, PDIM, N_SEQ, 1.f, nullptr, 0, true);      // Bt = V^T
  gemm(Ab, wosumT, mmb, N_SEQ, WDIM, PDIM, 1.f, nullptr, 0, true);   // masked_mh

  // --- cross attention (enc = x) ---
  gemm(mmb, wqcT, qb, N_SEQ, PDIM, WDIM, 1.f, nullptr, 0, true);
  gemm(xb, wkcT, kb, N_SEQ, PDIM, WDIM, 1.f, nullptr, 0, true);
  gemm(xb, wvcT, vb, N_SEQ, PDIM, WDIM, 1.f, nullptr, 0, true);
  transpose_bf16<<<(N_SEQ * PDIM) / 256, 256, 0, stream>>>(vb, vTb, N_SEQ, PDIM);
  gemm(qb, kb, Sb, N_SEQ, N_SEQ, PDIM, invsqrtP, nullptr, 0, true);
  softmax_rows_bf16<<<N_SEQ, 256, 0, stream>>>(Sb, N_SEQ);
  gemm(Sb, vTb, Ab, N_SEQ, PDIM, N_SEQ, 1.f, nullptr, 0, true);
  gemm(Ab, wosumT, mhf, N_SEQ, WDIM, PDIM, 1.f, nullptr, 0, false);  // mh (f32)

  // --- residual + LN ---
  add_ln_kernel<<<N_SEQ, 256, 0, stream>>>(mhf, x, lng, lnb, hf, hb);

  // --- FFN ---
  gemm(hb, w1T, y1b, N_SEQ, FDIM, WDIM, 1.f, b1, 1, true);   // + bias, LeakyReLU
  gemm(y1b, w2T, y2f, N_SEQ, WDIM, FDIM, 1.f, b2, 0, false);

  // --- final residual + LN -> d_out ---
  add_ln_kernel<<<N_SEQ, 256, 0, stream>>>(y2f, hf, lng, lnb, outp, nullptr);
}
